// RelationNetwork_54528904790573
// MI455X (gfx1250) — matrix operand order repaired and checked by reading, not compiled, not run on hardware
//
#include <hip/hip_runtime.h>
#include <hip/hip_bf16.h>

#define B_   32
#define N_   64
#define D_   128
#define T_   256
#define OUT_ 128
#define EPS_ 1e-5f

typedef __attribute__((ext_vector_type(16))) __bf16 v16bf;
typedef __attribute__((ext_vector_type(8)))  __bf16 v8bf;
typedef __attribute__((ext_vector_type(8)))  float  v8f;

// ---------------------------------------------------------------------------
// XOR-swizzled LDS addressing: 16B blocks (8 bf16) permuted by row so the
// A-tile b128 loads (16 lanes, 16 consecutive rows, same block) hit 16
// distinct 4-bank groups => conflict-free with zero padding (64KB exactly).
__device__ __forceinline__ int swz(int row, int col) {
    int blk = col >> 3;
    blk = (blk & 16) | ((blk ^ row) & 15);
    return row * 256 + blk * 8 + (col & 7);
}

// ---------------------------------------------------------------------------
// Pack g_W (3,T,T) fp32 -> bf16 in WMMA B-operand order (layers 1,2 used):
// dest[((l*8+kt)*16+nt)*32 + L][kk] = W[l][nt*16+(L&15)][kt*32+(L>>4)*16+kk]
__global__ void pack_gW(const float* __restrict__ gW, __bf16* __restrict__ wp) {
    int idx = blockIdx.x * blockDim.x + threadIdx.x;
    if (idx >= 3 * T_ * T_) return;
    int l  = idx / (T_ * T_);
    int r  = idx % (T_ * T_);
    int kk = r & 15;
    int L  = (r >> 4) & 31;
    int nt = (r >> 9) & 15;
    int kt = (r >> 13) & 7;
    int n  = nt * 16 + (L & 15);
    int t  = kt * 32 + (L >> 4) * 16 + kk;
    wp[idx] = (__bf16)gW[(l * T_ + n) * T_ + t];
}

// ---------------------------------------------------------------------------
// Layer-1 decomposition: pair row is [x_i || x_j], so
//   G-layer0(pair) = (x_i @ Wl^T + x_j @ Wr^T + b) -> BN -> ReLU
// Precompute per OBJECT (2048 rows, not 131072 pair rows):
//   XA'[b,i,u] = (x_i . W0[u,:128]) * A0[u] + C0[u]     (affine folded here)
//   XC'[b,j,u] = (x_j . W0[u,128:]) * A0[u]
// where A0 = gamma*rsqrt(var+eps), C0 = (bias-mean)*A0 + beta.
__global__ void g0_halves(const float* __restrict__ x,
                          const float* __restrict__ gW,
                          const float* __restrict__ g_b,
                          const float* __restrict__ g_gamma,
                          const float* __restrict__ g_beta,
                          const float* __restrict__ g_mean,
                          const float* __restrict__ g_var,
                          float* __restrict__ XA, float* __restrict__ XC) {
    int idx = blockIdx.x * blockDim.x + threadIdx.x;   // over 2*B*N*T
    if (idx >= 2 * B_ * N_ * T_) return;
    int which = (idx >= B_ * N_ * T_) ? 1 : 0;         // 0 = left, 1 = right
    int r   = idx - which * B_ * N_ * T_;
    int u   = r & (T_ - 1);
    int row = r >> 8;                                  // b*64 + obj
    const float* xr = x + row * D_;
    const float* wr = gW + u * T_ + which * D_;
    float s = 0.f;
#pragma unroll 4
    for (int t = 0; t < D_; ++t) s += xr[t] * wr[t];
    float rs = rsqrtf(g_var[u] + EPS_);
    float A0 = g_gamma[u] * rs;
    float C0 = (g_b[u] - g_mean[u]) * A0 + g_beta[u];
    if (which) XC[r] = s * A0;
    else       XA[r] = s * A0 + C0;
}

// ---------------------------------------------------------------------------
// G main: per (batch, 128-pair-row tile): h1 = relu(XA'[i]+XC'[j]) -> LDS
// (bf16, swizzled), then 2x [GEMM(128x256 x 256x256) -> BN -> ReLU] via
// v_wmma_f32_16x16x32_bf16; layer-2 epilogue reduces rows -> partial sums.
__global__ __launch_bounds__(512)
void rn_g_kernel(const float* __restrict__ XA,
                 const float* __restrict__ XC,
                 const __bf16* __restrict__ wp,
                 const float* __restrict__ g_b,
                 const float* __restrict__ g_gamma,
                 const float* __restrict__ g_beta,
                 const float* __restrict__ g_mean,
                 const float* __restrict__ g_var,
                 float* __restrict__ partial) {
    __shared__ __bf16 sH[128 * 256];   // 64 KB, xor-swizzled

    const int b      = blockIdx.x >> 5;
    const int tile   = blockIdx.x & 31;
    const int tid    = threadIdx.x;
    const int w      = tid >> 5;       // wave id = output column tile (N)
    const int L      = tid & 31;
    const int half   = L >> 4;
    const int lane16 = L & 15;
    const int colu   = w * 16 + lane16;

    // ---- layer-1 activations from decomposed halves
    const float* XAb = XA + b * N_ * T_;
    const float* XCb = XC + b * N_ * T_;
    const int i0 = tile * 2;
    for (int idx = tid; idx < 128 * 256; idx += 512) {
        int row = idx >> 8, col = idx & 255;
        float z = XAb[(i0 + (row >> 6)) * T_ + col] + XCb[(row & 63) * T_ + col];
        sH[swz(row, col)] = (__bf16)fmaxf(z, 0.f);
    }
    __syncthreads();

    for (int l = 1; l <= 2; ++l) {
        v8f acc[8];
#pragma unroll
        for (int rt = 0; rt < 8; ++rt)
            acc[rt] = (v8f){0.f, 0.f, 0.f, 0.f, 0.f, 0.f, 0.f, 0.f};

        const v16bf* wB = (const v16bf*)(wp + l * T_ * T_);
#pragma unroll
        for (int kt = 0; kt < 8; ++kt) {
            if (kt + 1 < 8)  // prefetch next B tile (keep near)
                __builtin_prefetch((const void*)&wB[((kt + 1) * 16 + w) * 32 + L], 0, 3);
            v16bf bt = wB[(kt * 16 + w) * 32 + L];
#pragma unroll
            for (int rt = 0; rt < 8; ++rt) {
                int row   = rt * 16 + lane16;
                int cbase = kt * 32 + half * 16;
                v8bf alo = *(const v8bf*)&sH[swz(row, cbase)];
                v8bf ahi = *(const v8bf*)&sH[swz(row, cbase + 8)];
                v16bf a;
#pragma unroll
                for (int q = 0; q < 8; ++q) { a[q] = alo[q]; a[8 + q] = ahi[q]; }
                acc[rt] = __builtin_amdgcn_wmma_f32_16x16x32_bf16(
                    false, a, false, bt, (short)0, acc[rt], false, false);
            }
        }

        // fused bias + BatchNorm(eval): z*A + C, then ReLU
        float rs = rsqrtf(g_var[l * T_ + colu] + EPS_);
        float A  = g_gamma[l * T_ + colu] * rs;
        float C  = (g_b[l * T_ + colu] - g_mean[l * T_ + colu]) * A
                   + g_beta[l * T_ + colu];

        if (l == 1) {
            __syncthreads();               // all reads of sH done
#pragma unroll
            for (int rt = 0; rt < 8; ++rt)
#pragma unroll
                for (int r = 0; r < 8; ++r) {
                    float v = fmaxf(acc[rt][r] * A + C, 0.f);
                    int row = rt * 16 + r + half * 8;
                    sH[swz(row, colu)] = (__bf16)v;
                }
            __syncthreads();               // all writes visible
        } else {
            float s = 0.f;
#pragma unroll
            for (int rt = 0; rt < 8; ++rt)
#pragma unroll
                for (int r = 0; r < 8; ++r)
                    s += fmaxf(acc[rt][r] * A + C, 0.f);
            s += __shfl_xor(s, 16, 32);    // combine lane halves (same column)
            if (half == 0)
                partial[(b * 32 + tile) * T_ + colu] = s;
        }
    }
}

// ---------------------------------------------------------------------------
__global__ void reduce_partial(const float* __restrict__ partial,
                               float* __restrict__ gout) {
    int idx = blockIdx.x * blockDim.x + threadIdx.x;
    if (idx >= B_ * T_) return;
    int b = idx / T_, u = idx % T_;
    float s = 0.f;
    for (int t = 0; t < 32; ++t) s += partial[(b * 32 + t) * T_ + u];
    gout[idx] = s;
}

// ---------------------------------------------------------------------------
// F: (32,256) -> 2x Linear+BN+ReLU -> Linear(256->128). ~10.5 MFLOP: one block.
__global__ __launch_bounds__(256)
void rn_f_kernel(const float* __restrict__ gout,
                 const float* __restrict__ fW,  const float* __restrict__ fb,
                 const float* __restrict__ fg,  const float* __restrict__ fbe,
                 const float* __restrict__ fm,  const float* __restrict__ fv,
                 const float* __restrict__ foW, const float* __restrict__ fob,
                 float* __restrict__ out) {
    __shared__ float h[2][B_ * T_];
    int tid = threadIdx.x;
    for (int idx = tid; idx < B_ * T_; idx += 256) h[0][idx] = gout[idx];
    __syncthreads();
    int buf = 0;
    for (int l = 0; l < 2; ++l) {
        for (int idx = tid; idx < B_ * T_; idx += 256) {
            int row = idx >> 8, u = idx & 255;
            const float* wr = fW + (l * T_ + u) * T_;
            float z = 0.f;
            for (int t = 0; t < T_; ++t) z += h[buf][row * T_ + t] * wr[t];
            float rs = rsqrtf(fv[l * T_ + u] + EPS_);
            float A  = fg[l * T_ + u] * rs;
            float C  = (fb[l * T_ + u] - fm[l * T_ + u]) * A + fbe[l * T_ + u];
            h[buf ^ 1][idx] = fmaxf(z * A + C, 0.f);
        }
        buf ^= 1;
        __syncthreads();
    }
    for (int idx = tid; idx < B_ * OUT_; idx += 256) {
        int row = idx / OUT_, o = idx % OUT_;
        const float* wr = foW + o * T_;
        float z = 0.f;
        for (int t = 0; t < T_; ++t) z += h[buf][row * T_ + t] * wr[t];
        out[idx] = z + fob[o];
    }
}

// ---------------------------------------------------------------------------
extern "C" void kernel_launch(void* const* d_in, const int* in_sizes, int n_in,
                              void* d_out, int out_size, void* d_ws, size_t ws_size,
                              hipStream_t stream) {
    const float* x       = (const float*)d_in[0];
    const float* g_W     = (const float*)d_in[1];
    const float* g_b     = (const float*)d_in[2];
    const float* g_gamma = (const float*)d_in[3];
    const float* g_beta  = (const float*)d_in[4];
    const float* g_mean  = (const float*)d_in[5];
    const float* g_var   = (const float*)d_in[6];
    const float* f_W     = (const float*)d_in[7];
    const float* f_b     = (const float*)d_in[8];
    const float* f_gamma = (const float*)d_in[9];
    const float* f_beta  = (const float*)d_in[10];
    const float* f_mean  = (const float*)d_in[11];
    const float* f_var   = (const float*)d_in[12];
    const float* fo_W    = (const float*)d_in[13];
    const float* fo_b    = (const float*)d_in[14];

    char* ws = (char*)d_ws;
    size_t off = 0;
    __bf16* wp   = (__bf16*)(ws + off); off += (size_t)3 * T_ * T_ * 2;   // 384 KB
    float*  part = (float*)(ws + off);  off += (size_t)B_ * 32 * T_ * 4;  // 1 MB
    float*  gout = (float*)(ws + off);  off += (size_t)B_ * T_ * 4;       // 32 KB
    float*  XA   = (float*)(ws + off);  off += (size_t)B_ * N_ * T_ * 4;  // 2 MB
    float*  XC   = (float*)(ws + off);  off += (size_t)B_ * N_ * T_ * 4;  // 2 MB

    pack_gW<<<(3 * T_ * T_ + 255) / 256, 256, 0, stream>>>(g_W, wp);
    g0_halves<<<(2 * B_ * N_ * T_ + 255) / 256, 256, 0, stream>>>(
        x, g_W, g_b, g_gamma, g_beta, g_mean, g_var, XA, XC);
    rn_g_kernel<<<B_ * 32, 512, 0, stream>>>(XA, XC, wp, g_b, g_gamma, g_beta,
                                             g_mean, g_var, part);
    reduce_partial<<<(B_ * T_ + 255) / 256, 256, 0, stream>>>(part, gout);
    rn_f_kernel<<<1, 256, 0, stream>>>(gout, f_W, f_b, f_gamma, f_beta,
                                       f_mean, f_var, fo_W, fo_b, (float*)d_out);
}
